// LSTM_35768487641178
// MI455X (gfx1250) — matrix operand order repaired and checked by reading, not compiled, not run on hardware
//
#include <hip/hip_runtime.h>

typedef __bf16 v16bf __attribute__((ext_vector_type(16)));
typedef __bf16 v8bf  __attribute__((ext_vector_type(8)));
typedef float  v8f   __attribute__((ext_vector_type(8)));

#define TM 128
#define TN 128
#define TK 32

// ---------------------------------------------------------------------------
// fp32 -> bf16 cast (grid-stride)
// ---------------------------------------------------------------------------
__global__ void cast_f32_bf16(const float* __restrict__ src,
                              __bf16* __restrict__ dst, int n) {
    for (int i = blockIdx.x * blockDim.x + threadIdx.x; i < n;
         i += gridDim.x * blockDim.x) {
        dst[i] = (__bf16)src[i];
    }
}

// ---------------------------------------------------------------------------
// Async global -> LDS copy of one 128x32 bf16 A-tile and one 128x32 bf16
// B-tile. Each of the 256 threads issues 2+2 GLOBAL_LOAD_ASYNC_TO_LDS_B128
// (16 B/lane straight into LDS, tracked by ASYNCcnt, no VGPR staging).
// ---------------------------------------------------------------------------
__device__ __forceinline__ void async_b128(__bf16* lds, const __bf16* gptr) {
    unsigned lo = (unsigned)(uintptr_t)lds;  // low 32 bits = LDS byte address
    asm volatile("global_load_async_to_lds_b128 %0, %1, off" ::"v"(lo),
                 "v"((const void*)gptr)
                 : "memory");
}

__device__ __forceinline__ void issue_tile(const __bf16* __restrict__ A,
                                           const __bf16* __restrict__ W,
                                           __bf16* As, __bf16* Bs, int tid,
                                           int mBase, int nBase, int k0,
                                           int K) {
    #pragma unroll
    for (int j = 0; j < 2; j++) {
        int id = tid + 256 * j;
        int r  = id >> 2;
        int cc = (id & 3) * 8;
        async_b128(&As[r * TK + cc], &A[(size_t)(mBase + r) * K + k0 + cc]);
        async_b128(&Bs[r * TK + cc], &W[(size_t)(nBase + r) * K + k0 + cc]);
    }
}

// ---------------------------------------------------------------------------
// C[M,N] = A[M,K](bf16,row-major) * W[N,K](bf16,row-major)^T + bias[N]
// 128x128 tile / block of 256 threads (8 waves: 4 in M x 2 in N).
// Each wave: 32x64 of C = 2x4 fragments of v_wmma_f32_16x16x32_bf16.
// Double-buffered LDS, async tile fills pipelined one tile ahead.
// ---------------------------------------------------------------------------
__global__ __launch_bounds__(256)
void gemm_bf16_wmma(const __bf16* __restrict__ A, const __bf16* __restrict__ W,
                    const float* __restrict__ bias, float* __restrict__ C,
                    int M, int N, int K) {
    __shared__ __align__(16) __bf16 As[2][TM * TK];
    __shared__ __align__(16) __bf16 Bs[2][TN * TK];

    const int tid  = threadIdx.x;
    const int lane = tid & 31;
    const int wave = tid >> 5;   // 0..7
    const int mw   = wave & 3;   // wave row slot: rows 32*mw .. +31
    const int nw   = wave >> 2;  // wave col slot: cols 64*nw .. +63
    const int half = lane >> 4;  // 0 or 1 (wave32 half)
    const int l16  = lane & 15;

    const int mBase = blockIdx.y * TM;
    const int nBase = blockIdx.x * TN;

    v8f acc[2][4];
    #pragma unroll
    for (int i = 0; i < 2; i++)
        #pragma unroll
        for (int j = 0; j < 4; j++)
            #pragma unroll
            for (int e = 0; e < 8; e++) acc[i][j][e] = 0.0f;

    // prologue: async-fill tile 0
    issue_tile(A, W, As[0], Bs[0], tid, mBase, nBase, 0, K);

    const int nK = K / TK;
    for (int kt = 0; kt < nK; kt++) {
        const int buf = kt & 1;
        if (kt + 1 < nK) {
            // async-fill next tile into the other buffer, then wait for the
            // current tile's 4 async ops (in-order completion => cnt <= 4).
            issue_tile(A, W, As[buf ^ 1], Bs[buf ^ 1], tid, mBase, nBase,
                       (kt + 1) * TK, K);
            asm volatile("s_wait_asynccnt 4" ::: "memory");
        } else {
            asm volatile("s_wait_asynccnt 0" ::: "memory");
        }
        __syncthreads();  // all waves' tile-kt data resident in LDS

        // --- load fragments per ISA wave32 layout ---
        v16bf af[2], bfv[4];
        #pragma unroll
        for (int sm = 0; sm < 2; sm++) {
            // A 16x32: lane L holds row (L&15); K chunks [8*half..], [16+8*half..]
            const __bf16* p = &As[buf][(mw * 32 + sm * 16 + l16) * TK];
            v8bf lo = *(const v8bf*)(p + 8 * half);
            v8bf hi = *(const v8bf*)(p + 16 + 8 * half);
            af[sm] = __builtin_shufflevector(lo, hi, 0, 1, 2, 3, 4, 5, 6, 7, 8,
                                             9, 10, 11, 12, 13, 14, 15);
        }
        #pragma unroll
        for (int sn = 0; sn < 4; sn++) {
            // B 32x16: lane L holds column (L&15); K chunks [8*half..], [16+8*half..]
            const __bf16* p =
                &Bs[buf][(nw * 64 + sn * 16 + l16) * TK];
            v8bf lo = *(const v8bf*)(p + 8 * half);
            v8bf hi = *(const v8bf*)(p + 16 + 8 * half);
            bfv[sn] = __builtin_shufflevector(lo, hi, 0, 1, 2, 3, 4, 5, 6, 7, 8,
                                              9, 10, 11, 12, 13, 14, 15);
        }

        #pragma unroll
        for (int sm = 0; sm < 2; sm++)
            #pragma unroll
            for (int sn = 0; sn < 4; sn++)
                acc[sm][sn] = __builtin_amdgcn_wmma_f32_16x16x32_bf16(
                    false, af[sm], false, bfv[sn], (short)0, acc[sm][sn], false,
                    false);

        __syncthreads();  // all reads of buf done before it is refilled
    }

    // --- epilogue: C VGPR r -> M = r + 8*half, N = lane&15 ---
    #pragma unroll
    for (int sm = 0; sm < 2; sm++)
        #pragma unroll
        for (int sn = 0; sn < 4; sn++) {
            int col = nBase + nw * 64 + sn * 16 + l16;
            float bv = bias ? bias[col] : 0.0f;
            int m0 = mBase + mw * 32 + sm * 16 + 8 * half;
            #pragma unroll
            for (int r = 0; r < 8; r++)
                C[(size_t)(m0 + r) * N + col] = acc[sm][sn][r] + bv;
        }
}

// ---------------------------------------------------------------------------
// Row-wise LayerNorm + gate combine. One block per batch row; H = 2048.
// gate = LN(zx)*ax + bx + LN(zh)*ah + bh    (std is ddof=1, eps on sigma)
// mode 0: out1 = sigmoid(gate)                                (f, i gates)
// mode 1: out1 = fg*c0 + ig*tanh(gate)                        (-> cx)
// mode 2: out1 = sigmoid(gate)*tanh(LN(cx)*ac + bc); out2=bf16(out1)  (-> hx)
// ---------------------------------------------------------------------------
__device__ __forceinline__ float block_sum(float v, float* red) {
    int t = threadIdx.x;
    red[t] = v;
    __syncthreads();
    for (int s = 128; s > 0; s >>= 1) {
        if (t < s) red[t] += red[t + s];
        __syncthreads();
    }
    float r = red[0];
    __syncthreads();
    return r;
}

__global__ __launch_bounds__(256)
void ln_combine(const float* __restrict__ Zx, const float* __restrict__ Zh,
                const float* __restrict__ axv, const float* __restrict__ bxv,
                const float* __restrict__ ahv, const float* __restrict__ bhv,
                int mode, const float* __restrict__ fg,
                const float* __restrict__ ig, const float* __restrict__ c0,
                const float* __restrict__ cxin, const float* __restrict__ acv,
                const float* __restrict__ bcv, float* __restrict__ out1,
                __bf16* __restrict__ out2) {
    const int Hn = 2048;
    __shared__ float red[256];
    const int t = threadIdx.x;
    const size_t base = (size_t)blockIdx.x * Hn;

    float zx[8], zh[8], zc[8];
    float sx = 0.f, sxx = 0.f, sh = 0.f, shh = 0.f, sc = 0.f, scc = 0.f;
    #pragma unroll
    for (int j = 0; j < 8; j++) {
        int c = t + 256 * j;
        float vx = Zx[base + c];
        float vh = Zh[base + c];
        zx[j] = vx; zh[j] = vh;
        sx += vx; sxx += vx * vx;
        sh += vh; shh += vh * vh;
        if (mode == 2) {
            float vc = cxin[base + c];
            zc[j] = vc;
            sc += vc; scc += vc * vc;
        } else {
            zc[j] = 0.f;
        }
    }
    sx = block_sum(sx, red);  sxx = block_sum(sxx, red);
    sh = block_sum(sh, red);  shh = block_sum(shh, red);
    const float mux = sx / Hn, muh = sh / Hn;
    const float vvx = fmaxf((sxx - Hn * mux * mux) / (Hn - 1), 0.f);
    const float vvh = fmaxf((shh - Hn * muh * muh) / (Hn - 1), 0.f);
    const float isx = 1.f / (sqrtf(vvx) + 1e-5f);
    const float ish = 1.f / (sqrtf(vvh) + 1e-5f);
    float muc = 0.f, isc = 0.f;
    if (mode == 2) {
        sc = block_sum(sc, red);
        scc = block_sum(scc, red);
        muc = sc / Hn;
        const float vvc = fmaxf((scc - Hn * muc * muc) / (Hn - 1), 0.f);
        isc = 1.f / (sqrtf(vvc) + 1e-5f);
    }

    #pragma unroll
    for (int j = 0; j < 8; j++) {
        int c = t + 256 * j;
        float g = (zx[j] - mux) * isx * axv[c] + bxv[c] +
                  (zh[j] - muh) * ish * ahv[c] + bhv[c];
        if (mode == 0) {
            out1[base + c] = 1.f / (1.f + __expf(-g));
        } else if (mode == 1) {
            float ct = tanhf(g);
            out1[base + c] = fg[base + c] * c0[base + c] + ig[base + c] * ct;
        } else {
            float o  = 1.f / (1.f + __expf(-g));
            float lc = (zc[j] - muc) * isc * acv[c] + bcv[c];
            float hx = o * tanhf(lc);
            out1[base + c] = hx;
            out2[base + c] = (__bf16)hx;
        }
    }
}

// ---------------------------------------------------------------------------
extern "C" void kernel_launch(void* const* d_in, const int* in_sizes, int n_in,
                              void* d_out, int out_size, void* d_ws,
                              size_t ws_size, hipStream_t stream) {
    const int B = 4096, IN = 1024, H = 2048, OUT = 1024;

    const float* inp = (const float*)d_in[0];
    const float* h0  = (const float*)d_in[1];
    const float* c0  = (const float*)d_in[2];
    const float* Wfh = (const float*)d_in[3];  const float* bfh = (const float*)d_in[4];
    const float* Wih = (const float*)d_in[5];  const float* bih = (const float*)d_in[6];
    const float* Wch = (const float*)d_in[7];  const float* bch = (const float*)d_in[8];
    const float* Woh = (const float*)d_in[9];  const float* boh = (const float*)d_in[10];
    const float* Wfx = (const float*)d_in[11]; const float* bfx = (const float*)d_in[12];
    const float* Wix = (const float*)d_in[13]; const float* bix = (const float*)d_in[14];
    const float* Wcx = (const float*)d_in[15]; const float* bcx = (const float*)d_in[16];
    const float* Wox = (const float*)d_in[17]; const float* box_ = (const float*)d_in[18];
    const float* Wdec = (const float*)d_in[19]; const float* bdec = (const float*)d_in[20];
    const float* ax = (const float*)d_in[21]; const float* bx = (const float*)d_in[22];
    const float* ah = (const float*)d_in[23]; const float* bh = (const float*)d_in[24];
    const float* ac = (const float*)d_in[25]; const float* bc = (const float*)d_in[26];

    // d_out layout: out[B,OUT] | hx[B,H] | cx[B,H]
    float* out_p = (float*)d_out;
    float* hx_p  = out_p + (size_t)B * OUT;
    float* cx_p  = hx_p + (size_t)B * H;

    // workspace carve-out
    char* ws = (char*)d_ws;
    size_t off = 0;
    auto alloc = [&](size_t bytes) -> void* {
        off = (off + 255) & ~(size_t)255;
        void* p = ws + off;
        off += bytes;
        return p;
    };
    __bf16* inp_b  = (__bf16*)alloc((size_t)B * IN * 2);
    __bf16* h0_b   = (__bf16*)alloc((size_t)B * H * 2);
    __bf16* Wfh_b  = (__bf16*)alloc((size_t)H * H * 2);
    __bf16* Wih_b  = (__bf16*)alloc((size_t)H * H * 2);
    __bf16* Wch_b  = (__bf16*)alloc((size_t)H * H * 2);
    __bf16* Woh_b  = (__bf16*)alloc((size_t)H * H * 2);
    __bf16* Wfx_b  = (__bf16*)alloc((size_t)H * IN * 2);
    __bf16* Wix_b  = (__bf16*)alloc((size_t)H * IN * 2);
    __bf16* Wcx_b  = (__bf16*)alloc((size_t)H * IN * 2);
    __bf16* Wox_b  = (__bf16*)alloc((size_t)H * IN * 2);
    __bf16* Wdec_b = (__bf16*)alloc((size_t)OUT * H * 2);
    float*  Zx     = (float*)alloc((size_t)B * H * 4);
    float*  Zh     = (float*)alloc((size_t)B * H * 4);
    float*  gF     = (float*)alloc((size_t)B * H * 4);
    float*  gI     = (float*)alloc((size_t)B * H * 4);
    __bf16* hx_b   = (__bf16*)alloc((size_t)B * H * 2);
    (void)ws_size; (void)n_in; (void)in_sizes; (void)out_size;

    // 1) casts to bf16
    cast_f32_bf16<<<1024, 256, 0, stream>>>(inp, inp_b, B * IN);
    cast_f32_bf16<<<1024, 256, 0, stream>>>(h0, h0_b, B * H);
    cast_f32_bf16<<<1024, 256, 0, stream>>>(Wfh, Wfh_b, H * H);
    cast_f32_bf16<<<1024, 256, 0, stream>>>(Wih, Wih_b, H * H);
    cast_f32_bf16<<<1024, 256, 0, stream>>>(Wch, Wch_b, H * H);
    cast_f32_bf16<<<1024, 256, 0, stream>>>(Woh, Woh_b, H * H);
    cast_f32_bf16<<<1024, 256, 0, stream>>>(Wfx, Wfx_b, H * IN);
    cast_f32_bf16<<<1024, 256, 0, stream>>>(Wix, Wix_b, H * IN);
    cast_f32_bf16<<<1024, 256, 0, stream>>>(Wcx, Wcx_b, H * IN);
    cast_f32_bf16<<<1024, 256, 0, stream>>>(Wox, Wox_b, H * IN);
    cast_f32_bf16<<<1024, 256, 0, stream>>>(Wdec, Wdec_b, OUT * H);

    const dim3 blk(256);
    const dim3 gGate(H / TN, B / TM);   // (16, 32)
    const dim3 gDec(OUT / TN, B / TM);  // (8, 32)

    // 2) f gate
    gemm_bf16_wmma<<<gGate, blk, 0, stream>>>(inp_b, Wfx_b, bfx, Zx, B, H, IN);
    gemm_bf16_wmma<<<gGate, blk, 0, stream>>>(h0_b, Wfh_b, bfh, Zh, B, H, H);
    ln_combine<<<B, blk, 0, stream>>>(Zx, Zh, ax, bx, ah, bh, 0, nullptr,
                                      nullptr, nullptr, nullptr, nullptr,
                                      nullptr, gF, nullptr);
    // 3) i gate
    gemm_bf16_wmma<<<gGate, blk, 0, stream>>>(inp_b, Wix_b, bix, Zx, B, H, IN);
    gemm_bf16_wmma<<<gGate, blk, 0, stream>>>(h0_b, Wih_b, bih, Zh, B, H, H);
    ln_combine<<<B, blk, 0, stream>>>(Zx, Zh, ax, bx, ah, bh, 0, nullptr,
                                      nullptr, nullptr, nullptr, nullptr,
                                      nullptr, gI, nullptr);
    // 4) c~ gate fused with cx = f*c0 + i*tanh(.)
    gemm_bf16_wmma<<<gGate, blk, 0, stream>>>(inp_b, Wcx_b, bcx, Zx, B, H, IN);
    gemm_bf16_wmma<<<gGate, blk, 0, stream>>>(h0_b, Wch_b, bch, Zh, B, H, H);
    ln_combine<<<B, blk, 0, stream>>>(Zx, Zh, ax, bx, ah, bh, 1, gF, gI, c0,
                                      nullptr, nullptr, nullptr, cx_p, nullptr);
    // 5) o gate fused with hx = sigmoid(o)*tanh(LN(cx))
    gemm_bf16_wmma<<<gGate, blk, 0, stream>>>(inp_b, Wox_b, box_, Zx, B, H, IN);
    gemm_bf16_wmma<<<gGate, blk, 0, stream>>>(h0_b, Woh_b, boh, Zh, B, H, H);
    ln_combine<<<B, blk, 0, stream>>>(Zx, Zh, ax, bx, ah, bh, 2, nullptr,
                                      nullptr, nullptr, cx_p, ac, bc, hx_p,
                                      hx_b);
    // 6) decoder: out = hx @ Wdec^T + bdec
    gemm_bf16_wmma<<<gDec, blk, 0, stream>>>(hx_b, Wdec_b, bdec, out_p, B, OUT,
                                             H);
}
